// LocalGlobalAdaptiveFusion_18631568130346
// MI455X (gfx1250) — compile-verified, matrix operand order checked by reading, not run on hardware
//
#include <hip/hip_runtime.h>
#include <hip/hip_bf16.h>
#include <math.h>

#define HDIM 512
#define BGRP 512
#define GIN  1028
#define KPAD 1056      // 33 * 32 (K padded to WMMA k-tile)
#define KT   33
#define H2   256

typedef __attribute__((ext_vector_type(16))) _Float16 v16h;
typedef __attribute__((ext_vector_type(8)))  _Float16 v8h;
typedef __attribute__((ext_vector_type(8)))  float    v8f;

// ---------------- workspace layout ----------------
// f32 region (element offsets):
#define OFF_COUNTS 0
#define OFF_NEDGES (OFF_COUNTS + BGRP)
#define OFF_SUMG   (OFF_NEDGES + BGRP)
#define OFF_SUMA   (OFF_SUMG + BGRP*HDIM)
#define OFF_H1     (OFF_SUMA + BGRP*HDIM)
#define OFF_H1R    (OFF_H1   + BGRP*HDIM)
#define OFF_WL     (OFF_H1R  + BGRP*HDIM)
#define OFF_WG     (OFF_WL   + BGRP)
#define F32_TOTAL  (OFF_WG   + BGRP)
// f16 region starts at byte F32_TOTAL*4 (32B aligned: 4202496 % 32 == 0)
#define GF16_ELEMS  (BGRP * KPAD)             // g matrix, row-major f16
#define W1F_ELEMS   (KT * 32 * 512)           // B-fragment-packed W1
#define ZERO_COUNT  (OFF_SUMA + BGRP*HDIM)    // counts+nedges+sumg+suma

// K index held by (lane, element e) of a 16-bit WMMA A/B fragment (ISA 7.12.2)
__device__ __forceinline__ int frag_k(int lane, int e) {
    int v = e >> 1, h = e & 1;
    int base = (v < 4) ? ((lane & 16) ? 8 : 0) : ((lane & 16) ? 24 : 16);
    return base + ((v & 3) << 1) + h;
}

// ---------------- kernels ----------------

__global__ void zero_ws(float* ws, int n) {
    int i = blockIdx.x * blockDim.x + threadIdx.x;
    if (i < n) ws[i] = 0.f;
}

// Sorted-segment pooling: each block owns a contiguous chunk of nodes,
// accumulates per-feature sums in registers, flushes with atomics only at
// graph boundaries. blockDim = 512 (one thread per feature).
#define POOL_CHUNK 256
__global__ __launch_bounds__(512)
void pool_kernel(const float* __restrict__ xg, const float* __restrict__ xa,
                 const int* __restrict__ batch, float* counts,
                 float* sumg, float* suma, int n) {
    int f = threadIdx.x;
    int start = blockIdx.x * POOL_CHUNK;
    if (start >= n) return;
    int end = min(start + POOL_CHUNK, n);
    int cur = batch[start];
    float accg = 0.f, acca = 0.f, cnt = 0.f;
    for (int i = start; i < end; ++i) {
        int b = batch[i];                       // uniform across block
        if (b != cur) {
            atomicAdd(&sumg[(size_t)cur * HDIM + f], accg);
            atomicAdd(&suma[(size_t)cur * HDIM + f], acca);
            if (f == 0) atomicAdd(&counts[cur], cnt);
            accg = 0.f; acca = 0.f; cnt = 0.f; cur = b;
        }
        accg += xg[(size_t)i * HDIM + f];
        acca += xa[(size_t)i * HDIM + f];
        cnt  += 1.f;
    }
    atomicAdd(&sumg[(size_t)cur * HDIM + f], accg);
    atomicAdd(&suma[(size_t)cur * HDIM + f], acca);
    if (f == 0) atomicAdd(&counts[cur], cnt);
}

// Same-graph edge counting via per-block LDS histogram (ds_add_f32).
__global__ __launch_bounds__(256)
void edge_kernel(const int* __restrict__ edge_index, const int* __restrict__ batch,
                 float* n_edges, int E) {
    __shared__ float hist[BGRP];
    for (int j = threadIdx.x; j < BGRP; j += 256) hist[j] = 0.f;
    __syncthreads();
    int stride = gridDim.x * blockDim.x;
    for (int e = blockIdx.x * blockDim.x + threadIdx.x; e < E; e += stride) {
        int src = edge_index[e];
        int dst = edge_index[E + e];
        int sg = batch[src], dg = batch[dst];
        if (sg == dg) atomicAdd(&hist[sg], 1.f);
    }
    __syncthreads();
    for (int j = threadIdx.x; j < BGRP; j += 256)
        if (hist[j] != 0.f) atomicAdd(&n_edges[j], hist[j]);
}

// Per-graph finalize: segment means, cosine divergence, complexity features,
// emit g row (row-major f16, K padded to 1056).
__global__ __launch_bounds__(256)
void finalize_kernel(const float* counts, const float* n_edges,
                     const float* sumg, const float* suma,
                     _Float16* __restrict__ g) {
    int b = blockIdx.x, tid = threadIdx.x;
    __shared__ float sdot[256], sng[256], sna[256];
    __shared__ float comp[4];
    float cnt = counts[b];
    float safe = fmaxf(cnt, 1.f);
    float dot = 0.f, ng = 0.f, na = 0.f;
    _Float16* grow = g + (size_t)b * KPAD;
    for (int f = tid; f < HDIM; f += 256) {
        float mg = sumg[(size_t)b * HDIM + f] / safe;
        float ma = suma[(size_t)b * HDIM + f] / safe;
        grow[f]        = (_Float16)mg;
        grow[HDIM + f] = (_Float16)ma;
        dot += mg * ma; ng += mg * mg; na += ma * ma;
    }
    sdot[tid] = dot; sng[tid] = ng; sna[tid] = na;
    __syncthreads();
    for (int s = 128; s > 0; s >>= 1) {
        if (tid < s) { sdot[tid]+=sdot[tid+s]; sng[tid]+=sng[tid+s]; sna[tid]+=sna[tid+s]; }
        __syncthreads();
    }
    if (tid == 0) {
        float nna = fmaxf(sqrtf(sng[0]), 1e-8f);
        float nnb = fmaxf(sqrtf(sna[0]), 1e-8f);
        float cosv = sdot[0] / (nna * nnb);
        float ne = n_edges[b];
        comp[0] = logf(cnt + 1.f) / logf(501.f);                    // scale
        comp[1] = ne / (cnt * (cnt - 1.f) + 1e-8f);                  // density
        comp[2] = fminf((ne / (cnt + 1e-8f)) * 0.1f, 1.f);           // avg_degree_norm
        comp[3] = (1.f - cosv) * 0.5f;                               // divergence
    }
    __syncthreads();
    if (tid < KPAD - 2 * HDIM) {                                     // cols 1024..1055
        float v = (tid < 4) ? comp[tid] : 0.f;
        grow[2 * HDIM + tid] = (_Float16)v;
    }
}

// Repack W1 (1028x512 f32 row-major) into B-fragment-order f16 tiles so the
// GEMM loads one contiguous v16h (32B) per lane per k-step.
__global__ __launch_bounds__(256)
void pack_w1(const float* __restrict__ W1, _Float16* __restrict__ w1f, int total) {
    int idx = blockIdx.x * blockDim.x + threadIdx.x;
    if (idx >= total) return;
    int e    = idx & 15;
    int lane = (idx >> 4) & 31;
    int tile = idx >> 9;
    int kt = tile >> 5, nt = tile & 31;
    int n = nt * 16 + (lane & 15);
    int k = kt * 32 + frag_k(lane, e);
    float v = (k < GIN) ? W1[(size_t)k * HDIM + n] : 0.f;
    w1f[idx] = (_Float16)v;
}

// h1 = g(f16, row-major, 512x1056) @ W1f + b1  via v_wmma_f32_16x16x32_f16.
// One wave per 16x16 output tile; 8 waves per block; 1024 tiles total.
__global__ __launch_bounds__(256)
void gemm1_wmma(const _Float16* __restrict__ g, const _Float16* __restrict__ w1f,
                const float* __restrict__ b1, float* __restrict__ h1) {
    int wave = threadIdx.x >> 5;
    int lane = threadIdx.x & 31;
    int tile = blockIdx.x * 8 + wave;
    int mt = tile >> 5, nt = tile & 31;
    int m  = mt * 16 + (lane & 15);
    int kb = (lane & 16) ? 8 : 0;            // lower/upper lane K-base (ISA layout)
    const _Float16* arow  = g + (size_t)m * KPAD + kb;
    const _Float16* bbase = w1f + ((size_t)nt * 32 + lane) * 16;
    v8f acc = {};
    for (int kt = 0; kt < KT; ++kt) {
        v8h alo = *(const v8h*)(arow + kt * 32);        // K = kb..kb+7
        v8h ahi = *(const v8h*)(arow + kt * 32 + 16);   // K = kb+16..kb+23
        v16h a;
        #pragma unroll
        for (int i = 0; i < 8; ++i) { a[i] = alo[i]; a[8 + i] = ahi[i]; }
        v16h bm = *(const v16h*)(bbase + (size_t)kt * (32 * 32 * 16));
        acc = __builtin_amdgcn_wmma_f32_16x16x32_f16(false, a, false, bm,
                                                     (short)0, acc, false, false);
    }
    int mb = mt * 16 + ((lane & 16) ? 8 : 0);
    int n  = nt * 16 + (lane & 15);
    float bias = b1[n];
    #pragma unroll
    for (int r = 0; r < 8; ++r)
        h1[(size_t)(mb + r) * HDIM + n] = acc[r] + bias;   // C/D: VGPR r -> M = mb+r
}

// LayerNorm over H=512 then ReLU. One block (256 threads) per row.
__global__ __launch_bounds__(256)
void ln_relu(const float* __restrict__ h1, const float* __restrict__ lng,
             const float* __restrict__ lnb, float* __restrict__ out) {
    int r = blockIdx.x, tid = threadIdx.x;
    __shared__ float red[256];
    float a = h1[(size_t)r * HDIM + tid];
    float b = h1[(size_t)r * HDIM + tid + 256];
    red[tid] = a + b; __syncthreads();
    for (int s = 128; s > 0; s >>= 1) { if (tid < s) red[tid] += red[tid + s]; __syncthreads(); }
    float mu = red[0] * (1.f / HDIM);
    __syncthreads();
    float da = a - mu, db = b - mu;
    red[tid] = da * da + db * db; __syncthreads();
    for (int s = 128; s > 0; s >>= 1) { if (tid < s) red[tid] += red[tid + s]; __syncthreads(); }
    float inv = rsqrtf(red[0] * (1.f / HDIM) + 1e-5f);
    float o0 = da * inv * lng[tid]       + lnb[tid];
    float o1 = db * inv * lng[tid + 256] + lnb[tid + 256];
    out[(size_t)r * HDIM + tid]       = fmaxf(o0, 0.f);
    out[(size_t)r * HDIM + tid + 256] = fmaxf(o1, 0.f);
}

// Tail MLP per graph: h2 = relu(h1r@W2+b2) (256 wide), raw = h2@W3+b3, softmax.
__global__ __launch_bounds__(256)
void tail_kernel(const float* __restrict__ h1r, const float* __restrict__ W2,
                 const float* __restrict__ b2, const float* __restrict__ W3,
                 const float* __restrict__ b3, float* wl, float* wg) {
    int bg = blockIdx.x, tid = threadIdx.x;
    __shared__ float hrow[HDIM];
    __shared__ float r0[256], r1[256];
    for (int f = tid; f < HDIM; f += 256) hrow[f] = h1r[(size_t)bg * HDIM + f];
    __syncthreads();
    float d = 0.f;
    for (int k = 0; k < HDIM; ++k) d += hrow[k] * W2[(size_t)k * H2 + tid];
    float h2 = fmaxf(d + b2[tid], 0.f);
    r0[tid] = h2 * W3[tid * 2 + 0];
    r1[tid] = h2 * W3[tid * 2 + 1];
    __syncthreads();
    for (int s = 128; s > 0; s >>= 1) {
        if (tid < s) { r0[tid] += r0[tid + s]; r1[tid] += r1[tid + s]; }
        __syncthreads();
    }
    if (tid == 0) {
        float raw0 = r0[0] + b3[0], raw1 = r1[0] + b3[1];
        float mx = fmaxf(raw0, raw1);
        float e0 = __expf(raw0 - mx), e1 = __expf(raw1 - mx);
        float inv = 1.f / (e0 + e1);
        wl[bg] = e0 * inv;
        wg[bg] = e1 * inv;
    }
}

// Streaming fusion: one block per node, float4 vector traffic.
__global__ __launch_bounds__(128)
void fuse_kernel(const float* __restrict__ xg, const float* __restrict__ xa,
                 const int* __restrict__ batch, const float* __restrict__ wl,
                 const float* __restrict__ wg, float* __restrict__ out) {
    int node = blockIdx.x;
    int bg = batch[node];
    float a = wl[bg], c = wg[bg];
    const float4* g4 = (const float4*)(xg + (size_t)node * HDIM);
    const float4* p4 = (const float4*)(xa + (size_t)node * HDIM);
    float4*       o4 = (float4*)(out + (size_t)node * HDIM);
    float4 u = g4[threadIdx.x];
    float4 v = p4[threadIdx.x];
    float4 o;
    o.x = a * u.x + c * v.x; o.y = a * u.y + c * v.y;
    o.z = a * u.z + c * v.z; o.w = a * u.w + c * v.w;
    o4[threadIdx.x] = o;
}

// ---------------- launcher ----------------
extern "C" void kernel_launch(void* const* d_in, const int* in_sizes, int n_in,
                              void* d_out, int out_size, void* d_ws, size_t ws_size,
                              hipStream_t stream) {
    const float* xg    = (const float*)d_in[0];
    const float* xa    = (const float*)d_in[1];
    const int*   eidx  = (const int*)d_in[2];
    const int*   batch = (const int*)d_in[3];
    const float* W1    = (const float*)d_in[4];
    const float* b1    = (const float*)d_in[5];
    const float* lng   = (const float*)d_in[6];
    const float* lnb   = (const float*)d_in[7];
    const float* W2    = (const float*)d_in[8];
    const float* b2    = (const float*)d_in[9];
    const float* W3    = (const float*)d_in[10];
    const float* b3    = (const float*)d_in[11];
    float* out = (float*)d_out;

    int N = in_sizes[0] / HDIM;
    int E = in_sizes[2] / 2;

    float* ws      = (float*)d_ws;
    float* counts  = ws + OFF_COUNTS;
    float* nedges  = ws + OFF_NEDGES;
    float* sumg    = ws + OFF_SUMG;
    float* suma    = ws + OFF_SUMA;
    float* h1      = ws + OFF_H1;
    float* h1r     = ws + OFF_H1R;
    float* wl      = ws + OFF_WL;
    float* wg      = ws + OFF_WG;
    _Float16* gf16 = (_Float16*)(ws + F32_TOTAL);
    _Float16* w1f  = gf16 + GF16_ELEMS;

    zero_ws<<<(ZERO_COUNT + 255) / 256, 256, 0, stream>>>(ws, ZERO_COUNT);

    pool_kernel<<<(N + POOL_CHUNK - 1) / POOL_CHUNK, 512, 0, stream>>>(
        xg, xa, batch, counts, sumg, suma, N);

    edge_kernel<<<1024, 256, 0, stream>>>(eidx, batch, nedges, E);

    finalize_kernel<<<BGRP, 256, 0, stream>>>(counts, nedges, sumg, suma, gf16);

    pack_w1<<<(W1F_ELEMS + 255) / 256, 256, 0, stream>>>(W1, w1f, W1F_ELEMS);

    gemm1_wmma<<<(BGRP / 16) * (HDIM / 16) / 8, 256, 0, stream>>>(gf16, w1f, b1, h1);

    ln_relu<<<BGRP, 256, 0, stream>>>(h1, lng, lnb, h1r);

    tail_kernel<<<BGRP, 256, 0, stream>>>(h1r, W2, b2, W3, b3, wl, wg);

    fuse_kernel<<<N, 128, 0, stream>>>(xg, xa, batch, wl, wg, out);
}
